// MultiScaleSpectralAttention_17703855194215
// MI455X (gfx1250) — compile-verified, matrix operand order checked by reading, not run on hardware
//
#include <hip/hip_runtime.h>
#include <hip/hip_bf16.h>
#include <math.h>

// ---------------------------------------------------------------------------
// MultiScaleSpectralAttention for MI455X (gfx1250, wave32).
// B=8, N=4096, D=768, H=8, HD=96, WIN=8, FG=2049, FL=5.
// Memory-bound (~1GB traffic @ 23.3TB/s => ~45us floor). ctx-MLPs use
// v_wmma_f32_16x16x32_bf16; FFT runs in LDS (one HBM read+write per channel).
// ---------------------------------------------------------------------------

#define Bq 8
#define Nq 4096
#define Dq 768
#define Hq 8
#define HDq 96
#define FGq 2049
#define FLq 5

typedef __attribute__((ext_vector_type(16))) __bf16 v16bf;
typedef __attribute__((ext_vector_type(8)))  float  v8f;

__device__ __forceinline__ float gelu_exact(float x) {
    return 0.5f * x * (1.0f + erff(x * 0.70710678118654752f));
}

// ---------------------------------------------------------------------------
// Kernel 0: zero the ctx accumulator (must be re-zeroed every call).
// ---------------------------------------------------------------------------
__global__ void zero_f32(float* __restrict__ p, int n) {
    int i = blockIdx.x * 256 + threadIdx.x;
    if (i < n) p[i] = 0.0f;
}

// ---------------------------------------------------------------------------
// Kernel 1: LayerNorm + ctx partial sums. One block = 16 tokens of one batch.
// 256 threads, 3 columns each (768 = 3*256). Single-pass mean/var (sum,sumsq).
// ---------------------------------------------------------------------------
__global__ __launch_bounds__(256)
void ln_ctx_kernel(const float* __restrict__ x, const float* __restrict__ lnw,
                   const float* __restrict__ lnb, float* __restrict__ xn,
                   float* __restrict__ ctx)
{
    __shared__ float2 red[256];
    const int tid = threadIdx.x;
    const int b   = blockIdx.x / (Nq / 16);
    const int t0  = (blockIdx.x % (Nq / 16)) * 16;

    const float w0 = lnw[tid], w1 = lnw[tid + 256], w2 = lnw[tid + 512];
    const float c0 = lnb[tid], c1 = lnb[tid + 256], c2 = lnb[tid + 512];
    float acc0 = 0.f, acc1 = 0.f, acc2 = 0.f;

    for (int t = 0; t < 16; ++t) {
        const size_t base = ((size_t)(b * Nq + t0 + t)) * Dq;
        float v0 = x[base + tid], v1 = x[base + tid + 256], v2 = x[base + tid + 512];
        red[tid] = make_float2(v0 + v1 + v2, v0*v0 + v1*v1 + v2*v2);
        __syncthreads();
        for (int o = 128; o > 0; o >>= 1) {
            if (tid < o) {
                red[tid].x += red[tid + o].x;
                red[tid].y += red[tid + o].y;
            }
            __syncthreads();
        }
        const float mu   = red[0].x * (1.0f / Dq);
        const float var  = red[0].y * (1.0f / Dq) - mu * mu;
        const float rstd = rsqrtf(var + 1e-5f);
        __syncthreads();                       // red[0] consumed before reuse

        float y0 = (v0 - mu) * rstd * w0 + c0;
        float y1 = (v1 - mu) * rstd * w1 + c1;
        float y2 = (v2 - mu) * rstd * w2 + c2;
        xn[base + tid]       = y0;
        xn[base + tid + 256] = y1;
        xn[base + tid + 512] = y2;
        acc0 += y0; acc1 += y1; acc2 += y2;
    }
    atomicAdd(&ctx[b * Dq + tid],       acc0);
    atomicAdd(&ctx[b * Dq + tid + 256], acc1);
    atomicAdd(&ctx[b * Dq + tid + 512], acc2);
}

// ---------------------------------------------------------------------------
// Kernel 2: generic WMMA GEMM  out[8 x Ncols] = act(ascale*A[8xK] @ W[KxNcols] + bias)
// One wave per 16-column tile; M padded 8->16 (rows 8..15 duplicated, unused).
// A-frag (16-bit A 16x32): lane m = lane&15; VGPR v holds K = 16*(v/4) +
// 8*(lane>=16) + 2*(v%4) + {0,1}.  B-frag mirrored (lane = column).
// Emits v_wmma_f32_16x16x32_bf16 + global_prefetch_b8.
// ---------------------------------------------------------------------------
__global__ __launch_bounds__(32)
void gemm_wmma(const float* __restrict__ A, float ascale,
               const float* __restrict__ W, const float* __restrict__ bias,
               float* __restrict__ out, int K, int Ncols, int act)
{
    const int lane  = threadIdx.x;
    const int m8    = lane & 7;                    // rows 0..7 (dup to 8..15)
    const int nIdx  = lane & 15;
    const int col   = blockIdx.x * 16 + nIdx;
    const int colC  = (col < Ncols) ? col : (Ncols - 1);
    const int hi8   = (lane >> 4) << 3;            // 0 or 8

    v8f acc = {};
    for (int k0 = 0; k0 < K; k0 += 32) {
        v16bf a, b;
#pragma unroll
        for (int v = 0; v < 8; ++v) {
            const int kk = k0 + ((v >> 2) << 4) + hi8 + ((v & 3) << 1);
            a[2*v]   = (__bf16)(A[m8 * K + kk]     * ascale);
            a[2*v+1] = (__bf16)(A[m8 * K + kk + 1] * ascale);
            b[2*v]   = (__bf16)(W[(size_t)kk       * Ncols + colC]);
            b[2*v+1] = (__bf16)(W[(size_t)(kk + 1) * Ncols + colC]);
        }
        if (k0 + 32 < K)   // stream next weight slab (global_prefetch_b8)
            __builtin_prefetch(&W[(size_t)(k0 + 32) * Ncols + colC], 0, 1);
        acc = __builtin_amdgcn_wmma_f32_16x16x32_bf16(
                  false, a, false, b, (short)0, acc, false, false);
    }

    if (lane < 16 && col < Ncols) {
        const float bv = bias[colC];
#pragma unroll
        for (int r = 0; r < 8; ++r) {              // C: VGPR r @ lanes0-15 = row r
            float v = acc[r] + bv;
            if (act == 1)      v = gelu_exact(v);
            else if (act == 2) v = 1.0f / (1.0f + __expf(-v));
            out[(size_t)r * Ncols + col] = v;
        }
    }
}

// ---------------------------------------------------------------------------
// In-LDS radix-2 DIT complex FFT, 4096 points, 256 threads.
// ---------------------------------------------------------------------------
__device__ void fft4096_lds(float* sr, float* si, int tid)
{
    __syncthreads();
    for (int n = tid; n < 4096; n += 256) {        // bit reversal (12 bits)
        int r = __brev((unsigned)n) >> 20;
        if (r > n) {
            float a = sr[n]; sr[n] = sr[r]; sr[r] = a;
            float b = si[n]; si[n] = si[r]; si[r] = b;
        }
    }
    for (int s = 1; s <= 12; ++s) {
        __syncthreads();
        const int half = 1 << (s - 1);
        for (int t = tid; t < 2048; t += 256) {
            const int pos = t & (half - 1);
            const int i0  = ((t >> (s - 1)) << s) + pos;
            const int i1  = i0 + half;
            const float ang = -3.14159265358979f * (float)pos / (float)half;
            float c, sn;
            __sincosf(ang, &sn, &c);
            const float br = sr[i1], bi = si[i1];
            const float tr = br * c - bi * sn;
            const float ti = br * sn + bi * c;
            const float ar = sr[i0], ai = si[i0];
            sr[i0] = ar + tr; si[i0] = ai + ti;
            sr[i1] = ar - tr; si[i1] = ai - ti;
        }
    }
    __syncthreads();
}

// ---------------------------------------------------------------------------
// Kernel 3: global branch. One block per (b,h,channel-pair): pack channels
// (2c,2c+1) as real/imag, forward FFT, Hermitian unpack -> modulate+cgelu ->
// repack (conjugated), forward FFT again = inverse, write x_global (B,N,D).
// ---------------------------------------------------------------------------
__global__ __launch_bounds__(256)
void global_spectral_kernel(const float* __restrict__ xn,
                            const float* __restrict__ bf_g,
                            const float* __restrict__ bb_g,
                            const float* __restrict__ ag,
                            float* __restrict__ glob)
{
    __shared__ float sr[4096];
    __shared__ float si[4096];
    const int tid = threadIdx.x;
    const int b   = blockIdx.x / (Hq * 48);
    const int rem = blockIdx.x % (Hq * 48);
    const int h   = rem / 48;
    const int c   = rem % 48;
    const size_t chOff = (size_t)h * HDq + 2 * c;

    for (int n = tid; n < 4096; n += 256) {        // hann(4096): /(N-1)
        const float hn = 0.5f - 0.5f * __cosf((float)n * (6.2831853071795865f / 4095.0f));
        const float* p = xn + ((size_t)(b * Nq + n)) * Dq + chOff;
        sr[n] = p[0] * hn;
        si[n] = p[1] * hn;
    }
    fft4096_lds(sr, si, tid);

    const float s64 = 1.0f / 64.0f;                // ortho: 1/sqrt(4096)
    for (int k = tid; k < 2049; k += 256) {
        const int km  = (4096 - k) & 4095;
        const float zkr = sr[k],  zki = si[k];
        const float zmr = sr[km], zmi = si[km];
        // U = spectrum of even channel, V = of odd channel (Hermitian split)
        const float Ur = 0.5f * (zkr + zmr) * s64, Ui = 0.5f * (zki - zmi) * s64;
        const float Vr = 0.5f * (zki + zmi) * s64, Vi = 0.5f * (zmr - zkr) * s64;

        const size_t fi  = (size_t)h * FGq + k;
        const float* agp = ag + (((size_t)b * Hq + h) * FGq + k) * 2;
        const float ef = bf_g[fi] * (1.0f + agp[0]);
        const float eb = bb_g[fi] + agp[1];

        float zur = Ur * ef + eb, zui = Ui * ef;
        float zvr = Vr * ef + eb, zvi = Vi * ef;
        { float m = sqrtf(zur*zur + zui*zui); float g = gelu_exact(m) / (m + 1e-6f); zur *= g; zui *= g; }
        { float m = sqrtf(zvr*zvr + zvi*zvi); float g = gelu_exact(m) / (m + 1e-6f); zvr *= g; zvi *= g; }

        // W[k] = Yu + i*Yv ; store conj(W) so a forward FFT computes the IDFT
        sr[k] = zur - zvi;
        si[k] = -(zui + zvr);
        if (k > 0 && k < 2048) {                   // W[N-k] = conj(Yu)+i*conj(Yv)
            sr[4096 - k] = zur + zvi;
            si[4096 - k] = -(zvr - zui);
        }
    }
    fft4096_lds(sr, si, tid);

    for (int n = tid; n < 4096; n += 256) {        // conj + ortho 1/64
        float* p = glob + ((size_t)(b * Nq + n)) * Dq + chOff;
        p[0] =  sr[n] * s64;                       // even channel
        p[1] = -si[n] * s64;                       // odd channel
    }
}

// ---------------------------------------------------------------------------
// Kernel 4: local STFT branch (8-pt rDFT, hard-coded twiddles) + gate fusion
// + residual. One block per (b, window); 256 threads x 3 columns, coalesced.
// ---------------------------------------------------------------------------
__global__ __launch_bounds__(256)
void local_fuse_kernel(const float* __restrict__ x, const float* __restrict__ xn,
                       const float* __restrict__ glob,
                       const float* __restrict__ bf_l, const float* __restrict__ bb_l,
                       const float* __restrict__ al, const float* __restrict__ alpha,
                       float* __restrict__ out)
{
    const float C8[8] = { 1.f,  0.70710678f, 0.f, -0.70710678f,
                         -1.f, -0.70710678f, 0.f,  0.70710678f };
    const float S8[8] = { 0.f,  0.70710678f, 1.f,  0.70710678f,
                          0.f, -0.70710678f,-1.f, -0.70710678f };
    const int tid = threadIdx.x;
    const int b   = blockIdx.x / (Nq / 8);
    const int n0  = (blockIdx.x % (Nq / 8)) * 8;
    const float r8 = 0.35355339059327f;            // 1/sqrt(8) (ortho)

    for (int i = 0; i < 3; ++i) {
        const int d = tid + i * 256;
        const int h = d / HDq;
        float s[8];
#pragma unroll
        for (int j = 0; j < 8; ++j) {              // hann(8): /(WIN-1)
            const float hn = 0.5f - 0.5f * __cosf((float)j * (6.2831853071795865f / 7.0f));
            s[j] = xn[((size_t)(b * Nq + n0 + j)) * Dq + d] * hn;
        }
        float Yr[5], Yi[5];
#pragma unroll
        for (int k = 0; k < 5; ++k) {
            float ar = 0.f, ai = 0.f;
#pragma unroll
            for (int j = 0; j < 8; ++j) {
                const int m = (k * j) & 7;
                ar += s[j] * C8[m];
                ai -= s[j] * S8[m];
            }
            const int fi = h * FLq + k;
            const float* alp = al + (size_t)b * (Hq * FLq * 2) + fi * 2;
            const float ef = bf_l[fi] * (1.0f + alp[0]);
            const float eb = bb_l[fi] + alp[1];
            float zr = ar * r8 * ef + eb;
            float zi = ai * r8 * ef;
            const float m2 = sqrtf(zr * zr + zi * zi);
            const float g  = gelu_exact(m2) / (m2 + 1e-6f);
            Yr[k] = zr * g;  Yi[k] = zi * g;
        }
        const float a_bh = alpha[b * Hq + h];
#pragma unroll
        for (int j = 0; j < 8; ++j) {
            float y = Yr[0] + Yr[4] * C8[(4 * j) & 7];   // bins 0 & N/2 are real
#pragma unroll
            for (int k = 1; k < 4; ++k) {
                const int m = (k * j) & 7;
                y += 2.0f * (Yr[k] * C8[m] - Yi[k] * S8[m]);
            }
            y *= r8;
            const size_t idx = ((size_t)(b * Nq + n0 + j)) * Dq + d;
            out[idx] = x[idx] + a_bh * glob[idx] + (1.0f - a_bh) * y;
        }
    }
}

// ---------------------------------------------------------------------------
extern "C" void kernel_launch(void* const* d_in, const int* in_sizes, int n_in,
                              void* d_out, int out_size, void* d_ws, size_t ws_size,
                              hipStream_t stream)
{
    const float* x    = (const float*)d_in[0];
    const float* ln_w = (const float*)d_in[1];
    const float* ln_b = (const float*)d_in[2];
    const float* bf_g = (const float*)d_in[3];
    const float* bb_g = (const float*)d_in[4];
    const float* bf_l = (const float*)d_in[5];
    const float* bb_l = (const float*)d_in[6];
    const float* w1g  = (const float*)d_in[7];
    const float* b1g  = (const float*)d_in[8];
    const float* w2g  = (const float*)d_in[9];
    const float* b2g  = (const float*)d_in[10];
    const float* w1l  = (const float*)d_in[11];
    const float* b1l  = (const float*)d_in[12];
    const float* w2l  = (const float*)d_in[13];
    const float* b2l  = (const float*)d_in[14];
    const float* wf1  = (const float*)d_in[15];
    const float* bf1  = (const float*)d_in[16];
    const float* wf2  = (const float*)d_in[17];
    const float* bf2  = (const float*)d_in[18];
    float* out = (float*)d_out;

    // workspace layout
    float* xnbuf   = (float*)d_ws;                        // B*N*D
    float* globbuf = xnbuf   + (size_t)Bq * Nq * Dq;      // B*N*D
    float* ctx     = globbuf + (size_t)Bq * Nq * Dq;      // B*D (accumulator)
    float* h1g     = ctx  + Bq * Dq;                      // 8*768
    float* h1l     = h1g  + Bq * Dq;                      // 8*768
    float* hf      = h1l  + Bq * Dq;                      // 8*384
    float* agb     = hf   + Bq * (Dq / 2);                // 8*32784
    float* alb     = agb  + (size_t)Bq * Hq * FGq * 2;    // 8*80
    float* alphab  = alb  + Bq * Hq * FLq * 2;            // 8*8

    const int NCg = Hq * FGq * 2;   // 32784
    const int NCl = Hq * FLq * 2;   // 80

    zero_f32<<<(Bq * Dq + 255) / 256, 256, 0, stream>>>(ctx, Bq * Dq);
    ln_ctx_kernel<<<Bq * (Nq / 16), 256, 0, stream>>>(x, ln_w, ln_b, xnbuf, ctx);

    const float inv_n = 1.0f / (float)Nq;                 // ctx = mean over N
    gemm_wmma<<<Dq / 16, 32, 0, stream>>>(ctx, inv_n, w1g, b1g, h1g, Dq, Dq, 1);
    gemm_wmma<<<Dq / 16, 32, 0, stream>>>(ctx, inv_n, w1l, b1l, h1l, Dq, Dq, 1);
    gemm_wmma<<<(Dq / 2) / 16, 32, 0, stream>>>(ctx, inv_n, wf1, bf1, hf, Dq, Dq / 2, 1);
    gemm_wmma<<<(NCg + 15) / 16, 32, 0, stream>>>(h1g, 1.0f, w2g, b2g, agb, Dq, NCg, 0);
    gemm_wmma<<<(NCl + 15) / 16, 32, 0, stream>>>(h1l, 1.0f, w2l, b2l, alb, Dq, NCl, 0);
    gemm_wmma<<<1, 32, 0, stream>>>(hf, 1.0f, wf2, bf2, alphab, Dq / 2, Hq, 2);

    global_spectral_kernel<<<Bq * Hq * (HDq / 2), 256, 0, stream>>>(
        xnbuf, bf_g, bb_g, agb, globbuf);
    local_fuse_kernel<<<Bq * (Nq / 8), 256, 0, stream>>>(
        x, xnbuf, globbuf, bf_l, bb_l, alb, alphab, out);
}
